// ContrastiveLoss_36893769073136
// MI455X (gfx1250) — compile-verified, matrix operand order checked by reading, not run on hardware
//
#include <hip/hip_runtime.h>

typedef _Float16 v16h __attribute__((ext_vector_type(16)));
typedef _Float16 v8h  __attribute__((ext_vector_type(8)));
typedef float    v8f  __attribute__((ext_vector_type(8)));

#define B_ROWS  4096
#define NROWS   8192          // 2B
#define DDIM    256
#define NTILES  (NROWS / 16)  // 512 column tiles
#define KCHUNKS (DDIM / 32)   // 8 K-chunks of 32
#define TILE_HALFS (16 * DDIM) // halfs per 16-row tile (8 KB)

#define LOG2E 1.442695040888963f

// ---------------------------------------------------------------------------
// Kernel 1: row-normalize concat(first, second) and convert to f16.
// One 256-thread block per row; thread t owns column t.
// ---------------------------------------------------------------------------
__global__ void k_normalize(const float* __restrict__ first,
                            const float* __restrict__ second,
                            _Float16* __restrict__ zf16) {
  int row = blockIdx.x;
  int t   = threadIdx.x;
  const float* src = (row < B_ROWS) ? (first  + (size_t)row * DDIM)
                                    : (second + (size_t)(row - B_ROWS) * DDIM);
  float x  = src[t];
  float ss = x * x;
  #pragma unroll
  for (int m = 1; m < 32; m <<= 1) ss += __shfl_xor(ss, m, 32);
  __shared__ float lds[8];
  if ((t & 31) == 0) lds[t >> 5] = ss;
  __syncthreads();
  float tot = 0.f;
  #pragma unroll
  for (int i = 0; i < 8; ++i) tot += lds[i];
  float rn = 1.0f / sqrtf(tot);
  zf16[(size_t)row * DDIM + t] = (_Float16)(x * rn);
}

// ---------------------------------------------------------------------------
// Kernel 2: streaming masked logsumexp of sim = 2 * z z^T via WMMA.
// One wave per 16-row strip; A-panel register-resident; double-buffered
// B-panel registers; __launch_bounds__(256,1) raises the VGPR cap so the
// ~210 live VGPRs (A 64 + 2xB 64 + C 8 + s 8) stay in registers (no spill).
// sim <= 2.0 always (unit rows, T=0.5) -> fixed LSE shift of 2.0.
// ---------------------------------------------------------------------------
__device__ __forceinline__ void load_b(v16h (&b)[KCHUNKS],
                                       const _Float16* __restrict__ p) {
  #pragma unroll
  for (int kc = 0; kc < KCHUNKS; ++kc) {
    v8h lo = *(const v8h*)(p + kc * 32);
    v8h hi = *(const v8h*)(p + kc * 32 + 16);
    #pragma unroll
    for (int i = 0; i < 8; ++i) { b[kc][i] = lo[i]; b[kc][i + 8] = hi[i]; }
  }
}

__device__ __forceinline__ void tile_lse(const v16h (&a)[KCHUNKS],
                                         const v16h (&b)[KCHUNKS],
                                         float (&s)[8],
                                         int jt, int strip, int dsel) {
  v8f c = {0.f, 0.f, 0.f, 0.f, 0.f, 0.f, 0.f, 0.f};
  #pragma unroll
  for (int kc = 0; kc < KCHUNKS; ++kc)
    c = __builtin_amdgcn_wmma_f32_16x16x32_f16(false, a[kc], false, b[kc],
                                               (short)0, c, false, false);
  const bool dtile = (jt == strip);
  #pragma unroll
  for (int v = 0; v < 8; ++v) {
    // exp(2c - 2) = exp2(fma(c, 2*log2e, -2*log2e)): 1 fma + 1 v_exp_f32
    float e = __builtin_amdgcn_exp2f(fmaf(c[v], 2.0f * LOG2E, -2.0f * LOG2E));
    e = (dtile && (dsel == v)) ? 0.0f : e;   // exact diagonal mask
    s[v] += e;
  }
}

__global__ void __launch_bounds__(256, 1)
k_lse(const _Float16* __restrict__ z, float* __restrict__ denom) {
  const int lane  = threadIdx.x & 31;
  const int wid   = threadIdx.x >> 5;
  const int strip = blockIdx.x * 8 + wid;   // 0..511
  const int half  = lane >> 4;              // which 16-lane half
  const int m15   = lane & 15;

  // A operand: lane (m15) holds row strip*16+m15; halves split K ranges.
  const _Float16* pA = z + (size_t)(strip * 16 + m15) * DDIM + half * 8;
  v16h a[KCHUNKS];
  #pragma unroll
  for (int kc = 0; kc < KCHUNKS; ++kc) {
    v8h lo = *(const v8h*)(pA + kc * 32);
    v8h hi = *(const v8h*)(pA + kc * 32 + 16);
    #pragma unroll
    for (int i = 0; i < 8; ++i) { a[kc][i] = lo[i]; a[kc][i + 8] = hi[i]; }
  }

  float s[8];
  #pragma unroll
  for (int v = 0; v < 8; ++v) s[v] = 0.f;
  const int dsel = m15 - 8 * half;  // diagonal hits accumulator v == dsel

  // Per-lane B base pointer for tile 0; tile jt base = pB + jt*TILE_HALFS.
  const _Float16* pB = z + (size_t)m15 * DDIM + half * 8;

  // Software-pipelined sweep over all column tiles (double-buffered B).
  v16h b0[KCHUNKS], b1[KCHUNKS];
  load_b(b0, pB);
  for (int jt = 0; jt < NTILES; jt += 2) {
    load_b(b1, pB + TILE_HALFS);           // tile jt+1 (imm offset 8 KB)
    tile_lse(a, b0, s, jt, strip, dsel);
    load_b(b0, pB + 2 * TILE_HALFS);       // tile jt+2 (imm offset 16 KB);
                                           // final iter reads the 8 KB ws slack
    tile_lse(a, b1, s, jt + 1, strip, dsel);
    pB += 2 * TILE_HALFS;
  }

  // Butterfly-reduce across the 16 lanes of this half (masks < 16 stay in-half).
  #pragma unroll
  for (int v = 0; v < 8; ++v) {
    float t = s[v];
    #pragma unroll
    for (int m = 1; m < 16; m <<= 1) t += __shfl_xor(t, m, 32);
    s[v] = t;
  }
  if (m15 == 0) {
    #pragma unroll
    for (int v = 0; v < 8; ++v)
      denom[strip * 16 + half * 8 + v] = 2.0f + __logf(s[v]);
  }
}

// ---------------------------------------------------------------------------
// Kernel 3: positive-pair similarity; res[i] = denom[i] - 2*dot(z_i, z_j),
// j = (i + B) mod 2B. One wave per row, f32 accumulation.
// ---------------------------------------------------------------------------
__global__ void k_pos(const _Float16* __restrict__ z,
                      const float* __restrict__ denom,
                      float* __restrict__ res) {
  int lane = threadIdx.x & 31;
  int row  = blockIdx.x * 8 + (threadIdx.x >> 5);
  int j    = (row + B_ROWS) & (NROWS - 1);
  v8h xi = *(const v8h*)(z + (size_t)row * DDIM + lane * 8);
  v8h xj = *(const v8h*)(z + (size_t)j   * DDIM + lane * 8);
  float d = 0.f;
  #pragma unroll
  for (int i = 0; i < 8; ++i) d += (float)xi[i] * (float)xj[i];
  #pragma unroll
  for (int m = 1; m < 32; m <<= 1) d += __shfl_xor(d, m, 32);
  if (lane == 0) res[row] = denom[row] - 2.0f * d;
}

// ---------------------------------------------------------------------------
// Kernel 4: deterministic fixed-order mean over 8192 rows.
// ---------------------------------------------------------------------------
__global__ void k_reduce(const float* __restrict__ res, float* __restrict__ out) {
  __shared__ float lds[256];
  int t = threadIdx.x;
  float s = 0.f;
  #pragma unroll
  for (int k = 0; k < NROWS / 256; ++k) s += res[t + 256 * k];
  lds[t] = s;
  __syncthreads();
  for (int w = 128; w > 0; w >>= 1) {
    if (t < w) lds[t] += lds[t + w];
    __syncthreads();
  }
  if (t == 0) out[0] = lds[0] * (1.0f / NROWS);
}

// ---------------------------------------------------------------------------
extern "C" void kernel_launch(void* const* d_in, const int* in_sizes, int n_in,
                              void* d_out, int out_size, void* d_ws, size_t ws_size,
                              hipStream_t stream) {
  const float* first  = (const float*)d_in[0];
  const float* second = (const float*)d_in[1];
  char* ws = (char*)d_ws;
  _Float16* zf16 = (_Float16*)ws;                   // 4 MB + 8 KB prefetch slack
  size_t z_bytes = (size_t)NROWS * DDIM * 2 + (size_t)TILE_HALFS * 2;
  float* denom   = (float*)(ws + z_bytes);          // 32 KB
  float* res     = denom + NROWS;                   // 32 KB
  float* out     = (float*)d_out;

  k_normalize<<<NROWS,      256, 0, stream>>>(first, second, zf16);
  k_lse      <<<NTILES / 8, 256, 0, stream>>>(zf16, denom);
  k_pos      <<<NROWS / 8,  256, 0, stream>>>(zf16, denom, res);
  k_reduce   <<<1,          256, 0, stream>>>(res, out);
}
